// Attention_69483980914985
// MI455X (gfx1250) — compile-verified, matrix operand order checked by reading, not run on hardware
//
#include <hip/hip_runtime.h>
#include <hip/hip_bf16.h>
#include <math.h>

#define L_SEQ 2048
#define E_DIM 1024
#define H_DIM 1024
#define G4    4096
#define NB    32
#define SLAB_BYTES (8 * 32 * 1024)            // 8 tiles x 32KB packed bf16 per WG
#define DYN_LDS    (SLAB_BYTES + 2048 + 512 + 128)

typedef __attribute__((ext_vector_type(16))) __bf16          v16bf;
typedef __attribute__((ext_vector_type(16))) unsigned short  v16u;
typedef __attribute__((ext_vector_type(8)))  unsigned short  v8u;
typedef __attribute__((ext_vector_type(8)))  float           v8f;
typedef __attribute__((ext_vector_type(4)))  unsigned int    v4u;
typedef __attribute__((ext_vector_type(8)))  int             v8i;
typedef __attribute__((ext_vector_type(4)))  int             v4i;

// round-to-nearest-even f32 -> bf16 bits
__device__ __forceinline__ unsigned short f2bf(float f) {
  unsigned u = __builtin_bit_cast(unsigned, f);
  u += 0x7fffu + ((u >> 16) & 1u);
  return (unsigned short)(u >> 16);
}

// CDNA5 16-bit A/B fragment K-permutation (ISA 7.12.2):
// lanes 0-15: VGPR0..3 hold K=0..7, VGPR4..7 hold K=16..23; lanes 16-31 shifted by +8
__device__ __forceinline__ int kperm(int lane, int e) {
  return e + (e < 8 ? 0 : 8) + (lane < 16 ? 0 : 8);
}

__device__ __forceinline__ float sigm(float x) { return 1.0f / (1.0f + __expf(-x)); }

// ---------------------------------------------------------------------------
// Kernel 1: target mean embedding + the two "special" step embeddings + meta
// ---------------------------------------------------------------------------
__global__ void prep_kernel(const int* __restrict__ x, const int* ts, const int* te,
                            const float* __restrict__ emb,
                            float* target, float* e_l, float* e_r, int* meta) {
  int start = ts[0], end = te[0];
  int first_l = (start > 0) ? 0 : (end + 1);
  int first_r = (end < L_SEQ - 1) ? (L_SEQ - 1) : (start - 1);
  int tspec_r = L_SEQ - 1 - first_r;        // step index (in scan order) of special input
  float inv = 1.0f / (float)(end - start + 1);
  int tl = x[first_l], tr = x[first_r];
  for (int j = threadIdx.x; j < E_DIM; j += blockDim.x) {
    float acc = 0.f;
    for (int i = start; i <= end; ++i) acc += emb[(size_t)x[i] * E_DIM + j];
    target[j] = acc * inv;
    e_l[j]    = emb[(size_t)tl * E_DIM + j];
    e_r[j]    = emb[(size_t)tr * E_DIM + j];
  }
  if (threadIdx.x == 0) { meta[0] = first_l; meta[1] = tspec_r; }
}

// ---------------------------------------------------------------------------
// Kernel 2: g_base = w_ih@target + b_ih + b_hh ; g_spec = w_ih@e + b_ih + b_hh
// ---------------------------------------------------------------------------
__global__ void base_gates_kernel(const float* __restrict__ w_ih_l, const float* b_ih_l, const float* b_hh_l,
                                  const float* __restrict__ w_ih_r, const float* b_ih_r, const float* b_hh_r,
                                  const float* __restrict__ target,
                                  const float* __restrict__ e_l, const float* __restrict__ e_r,
                                  float* gbase_l, float* gspec_l, float* gbase_r, float* gspec_r) {
  __shared__ float rt[128], rs[128];
  int r = blockIdx.x, p = blockIdx.y;
  const float* w  = (p == 0 ? w_ih_l : w_ih_r) + (size_t)r * E_DIM;
  const float* ev = (p == 0 ? e_l : e_r);
  float at = 0.f, as = 0.f;
  for (int k = threadIdx.x; k < E_DIM; k += 128) {
    float wv = w[k];
    at += wv * target[k];
    as += wv * ev[k];
  }
  rt[threadIdx.x] = at; rs[threadIdx.x] = as;
  __syncthreads();
  for (int s = 64; s > 0; s >>= 1) {
    if (threadIdx.x < s) { rt[threadIdx.x] += rt[threadIdx.x + s]; rs[threadIdx.x] += rs[threadIdx.x + s]; }
    __syncthreads();
  }
  if (threadIdx.x == 0) {
    if (p == 0) {
      float b = b_ih_l[r] + b_hh_l[r];
      gbase_l[r] = rt[0] + b; gspec_l[r] = rs[0] + b;
    } else {
      float b = b_ih_r[r] + b_hh_r[r];
      gbase_r[r] = rt[0] + b; gspec_r[r] = rs[0] + b;
    }
  }
}

// ---------------------------------------------------------------------------
// Kernel 3: permute w_hh -> bf16 WMMA A-fragments, grouped contiguously per
// recurrence workgroup: slot = wg*8 + wave, wave -> (gate g = v&3, htl = v>>2),
// m-tile = g*64 + wg*2 + htl. Each WG slab = 8 tiles x 32KB = 256KB contiguous.
// ---------------------------------------------------------------------------
__global__ void pack_whh_kernel(const float* __restrict__ whh_l, const float* __restrict__ whh_r,
                                unsigned short* pk_l, unsigned short* pk_r) {
  size_t idx = (size_t)blockIdx.x * blockDim.x + threadIdx.x;   // < 4096*1024
  const float* w    = blockIdx.y == 0 ? whh_l : whh_r;
  unsigned short* o = blockIdx.y == 0 ? pk_l : pk_r;
  int e = idx & 15, lane = (idx >> 4) & 31, kc = (idx >> 9) & 31;
  int slot = (int)(idx >> 14);              // 0..255
  int wg = slot >> 3, v = slot & 7;
  int g = v & 3, htl = v >> 2;
  int mt = g * 64 + wg * 2 + htl;
  int m = mt * 16 + (lane & 15);
  int k = kc * 32 + kperm(lane, e);
  o[idx] = f2bf(w[(size_t)m * H_DIM + k]);
}

// ---------------------------------------------------------------------------
// Kernel 4: persistent sequential LSTM pass.
// TDM loads this WG's 256KB W_hh fragment slab into LDS once; every step is
// then LDS-fed WMMA + pointwise + device-wide barrier.
// NOTE: B columns n!=0 intentionally carry garbage (every lane loads the same
// h chunk); WMMA columns are independent and we only read column 0.
// ---------------------------------------------------------------------------
__global__ void lstm_pass_kernel(const unsigned short* __restrict__ whh_pk,
                                 const float* __restrict__ gbase, const float* __restrict__ gspec,
                                 const int* __restrict__ meta, int meta_idx,
                                 float* __restrict__ hs_out,
                                 float* __restrict__ h_g, float* __restrict__ c_g,
                                 unsigned int* ctr, int init_from_global) {
  extern __shared__ char smem[];
  v16u*           slabW = (v16u*)smem;                                   // 256KB
  unsigned short* h_bf  = (unsigned short*)(smem + SLAB_BYTES);          // 2KB
  float*          gval  = (float*)(smem + SLAB_BYTES + 2048);            // 4*32 f32
  float*          c_l   = (float*)(smem + SLAB_BYTES + 2048 + 512);      // 32 f32

  const int tid = threadIdx.x, lane = tid & 31, wave = tid >> 5;
  const int g = wave & 3, htl = wave >> 2;
  const int tspec = meta[meta_idx];
  if (tid < 32) c_l[tid] = init_from_global ? c_g[blockIdx.x * 32 + tid] : 0.0f;

  // ---- TDM: DMA this WG's contiguous 256KB fragment slab into LDS ----
  if (wave == 0) {
    unsigned long long ga =
        (unsigned long long)(uintptr_t)whh_pk + (unsigned long long)blockIdx.x * SLAB_BYTES;
    unsigned lds_base = (unsigned)(uintptr_t)smem;   // flat LDS addr low 32 bits = LDS offset
    v4u g0;
    g0[0] = 1u;                                       // count=1, user mode, no gather
    g0[1] = lds_base;                                 // D#.lds_addr
    g0[2] = (unsigned)(ga & 0xffffffffu);             // D#.global_addr[31:0]
    g0[3] = (unsigned)((ga >> 32) & 0x01ffffffu) | (2u << 30);  // addr[56:32] | type=2
    v8i g1;
    g1[0] = (int)(3u << 16);                          // data_size = 8 bytes
    g1[1] = (int)(0x8000u << 16);                     // tensor_dim0 = 32768 (lo16)
    g1[2] = (int)(1u << 16);                          // tensor_dim0 hi = 0, tensor_dim1 = 1
    g1[3] = (int)(0x8000u << 16);                     // tile_dim0 = 32768 (1-D tile)
    g1[4] = 0;                                        // tile_dim1 = tile_dim2 = 0 (unused)
    g1[5] = 32768;                                    // tensor_dim0_stride
    g1[6] = 0;
    g1[7] = 0;
    v4i z4 = {0, 0, 0, 0};
    v8i z8 = {0, 0, 0, 0, 0, 0, 0, 0};
    __builtin_amdgcn_tensor_load_to_lds(g0, g1, z4, z4, z8, 0);
    __builtin_amdgcn_s_wait_tensorcnt(0);
  }
  __syncthreads();

  const v16u* __restrict__ Aw = slabW + (size_t)wave * 1024;  // this wave's tile (32 kc x 32 lanes)
  // B-fragment source: lanes 0-15 need K=kc*32+{0..7,16..23}, lanes 16-31 +8.
  // Two contiguous 16B chunks at (kc*64) and (kc*64+32) bytes from bp.
  const v8u* __restrict__ bp = (const v8u*)(h_bf + (lane < 16 ? 0 : 8));

  for (int step = 0; step < L_SEQ; ++step) {
    // broadcast h (global, released through grid barrier) into LDS as packed bf16
    __builtin_amdgcn_fence(__ATOMIC_ACQUIRE, "agent");
    {
      unsigned* h32 = (unsigned*)h_bf;
      for (int i = tid; i < H_DIM / 2; i += 256) {
        float2 hv = ((const float2*)h_g)[i];
        h32[i] = (unsigned)f2bf(hv.x) | ((unsigned)f2bf(hv.y) << 16);
      }
    }
    __syncthreads();

    // acc(16x16) = Whh_tile(16x1024) x B(1024x16); column 0 of B carries h
    v8f acc = {};
#pragma unroll 8
    for (int kc = 0; kc < 32; ++kc) {
      v16u au = Aw[kc * 32 + lane];
      v8u lo = bp[kc * 4];        // 16B: K = kc*32 + {0..7} (+8 for upper half-wave)
      v8u hi = bp[kc * 4 + 2];    // 16B: K = kc*32 + {16..23} (+8 for upper half-wave)
      v16u bu = __builtin_shufflevector(lo, hi, 0, 1, 2, 3, 4, 5, 6, 7,
                                        8, 9, 10, 11, 12, 13, 14, 15);
      acc = __builtin_amdgcn_wmma_f32_16x16x32_bf16(
          false, __builtin_bit_cast(v16bf, au),
          false, __builtin_bit_cast(v16bf, bu),
          (short)0, acc, false, false);
    }
    // column 0 of C/D lives in lane 0 (M=0..7) and lane 16 (M=8..15)
    if (lane == 0) {
#pragma unroll
      for (int i = 0; i < 8; ++i) gval[g * 32 + htl * 16 + i] = acc[i];
    } else if (lane == 16) {
#pragma unroll
      for (int i = 0; i < 8; ++i) gval[g * 32 + htl * 16 + 8 + i] = acc[i];
    }
    __syncthreads();

    // pointwise LSTM cell update for this WG's 32 h-indices
    if (tid < 32) {
      int jg = (int)blockIdx.x * 32 + tid;
      const float* base = (step == tspec) ? gspec : gbase;
      float gi = sigm (base[jg]             + gval[0 * 32 + tid]);
      float gf = sigm (base[H_DIM + jg]     + gval[1 * 32 + tid]);
      float gg = tanhf(base[2 * H_DIM + jg] + gval[2 * 32 + tid]);
      float go = sigm (base[3 * H_DIM + jg] + gval[3 * 32 + tid]);
      float c = gf * c_l[tid] + gi * gg;
      c_l[tid] = c;
      float h = go * tanhf(c);
      h_g[jg] = h;
      hs_out[(size_t)step * H_DIM + jg] = h;
    }
    __builtin_amdgcn_fence(__ATOMIC_RELEASE, "agent");
    __syncthreads();

    // grid-wide barrier: monotonic counter (graph-replay safe; memset per launch)
    if (tid == 0) {
      __hip_atomic_fetch_add(ctr, 1u, __ATOMIC_ACQ_REL, __HIP_MEMORY_SCOPE_AGENT);
      unsigned want = (unsigned)(NB * (step + 1));
      while (__hip_atomic_load(ctr, __ATOMIC_ACQUIRE, __HIP_MEMORY_SCOPE_AGENT) < want) {
        __builtin_amdgcn_s_sleep(1);
      }
    }
    __syncthreads();
  }
  if (tid < 32) c_g[blockIdx.x * 32 + tid] = c_l[tid];
}

// ---------------------------------------------------------------------------
// Kernel 5: prod = hs_l * hs_r (f32) AND bf16 A-fragment pack of prod
// ---------------------------------------------------------------------------
__global__ void prod_pack_kernel(const float* __restrict__ hs_l, const float* __restrict__ hs_r,
                                 float* __restrict__ prod, unsigned short* __restrict__ pA) {
  size_t idx = (size_t)blockIdx.x * blockDim.x + threadIdx.x;   // < 2048*1024
  int e = idx & 15, lane = (idx >> 4) & 31, kc = (idx >> 9) & 31, mt = (int)(idx >> 14);
  int m = mt * 16 + (lane & 15);
  int k = kc * 32 + kperm(lane, e);
  size_t off = (size_t)m * H_DIM + k;
  float v = hs_l[off] * hs_r[off];
  prod[off] = v;
  pA[idx] = f2bf(v);
}

// ---------------------------------------------------------------------------
// Kernel 6: lin1_w -> bf16 WMMA B-fragment order (B[k][n] = lin1_w[n][k])
// ---------------------------------------------------------------------------
__global__ void pack_lin1_kernel(const float* __restrict__ lin1_w, unsigned short* __restrict__ pB) {
  size_t idx = (size_t)blockIdx.x * blockDim.x + threadIdx.x;   // < 1024*1024
  int e = idx & 15, lane = (idx >> 4) & 31, kc = (idx >> 9) & 31, nt = (int)(idx >> 14);
  int n = nt * 16 + (lane & 15);
  int k = kc * 32 + kperm(lane, e);
  pB[idx] = f2bf(lin1_w[(size_t)n * H_DIM + k]);
}

// ---------------------------------------------------------------------------
// Kernel 7: t_out = tanh(prod @ lin1_w^T + b)  (2048x1024x1024 bf16 WMMA GEMM)
// ---------------------------------------------------------------------------
__global__ void gemm_tanh_kernel(const unsigned short* __restrict__ pA,
                                 const unsigned short* __restrict__ pB,
                                 const float* __restrict__ lin1_b, float* __restrict__ t_out) {
  const int tid = threadIdx.x, lane = tid & 31;
  const int wtile = (int)blockIdx.x * 8 + (tid >> 5);
  const int mt = wtile >> 6, nt = wtile & 63;
  const v16u* __restrict__ Af = (const v16u*)pA + (size_t)mt * 32 * 32;
  const v16u* __restrict__ Bf = (const v16u*)pB + (size_t)nt * 32 * 32;
  v8f acc = {};
#pragma unroll 8
  for (int kc = 0; kc < 32; ++kc) {
    if (kc < 24) {
      __builtin_prefetch(&Af[(kc + 8) * 32 + lane], 0, 1);
      __builtin_prefetch(&Bf[(kc + 8) * 32 + lane], 0, 1);
    }
    v16u au = Af[kc * 32 + lane];
    v16u bu = Bf[kc * 32 + lane];
    acc = __builtin_amdgcn_wmma_f32_16x16x32_bf16(
        false, __builtin_bit_cast(v16bf, au),
        false, __builtin_bit_cast(v16bf, bu),
        (short)0, acc, false, false);
  }
  int col = nt * 16 + (lane & 15);
  float bias = lin1_b[col];
  int rbase = mt * 16 + (lane < 16 ? 0 : 8);
#pragma unroll
  for (int i = 0; i < 8; ++i)
    t_out[(size_t)(rbase + i) * H_DIM + col] = tanhf(acc[i] + bias);
}

// ---------------------------------------------------------------------------
// Kernels 8-11: beta = t_out @ u ; softmax ; s = alfa @ prod ; out = lin2(s)
// ---------------------------------------------------------------------------
__global__ void beta_kernel(const float* __restrict__ t_out, const float* __restrict__ u,
                            float* __restrict__ beta) {
  __shared__ float red[256];
  int t = blockIdx.x;
  float a = 0.f;
  for (int j = threadIdx.x; j < H_DIM; j += 256) a += t_out[(size_t)t * H_DIM + j] * u[j];
  red[threadIdx.x] = a; __syncthreads();
  for (int s = 128; s > 0; s >>= 1) {
    if (threadIdx.x < s) red[threadIdx.x] += red[threadIdx.x + s];
    __syncthreads();
  }
  if (threadIdx.x == 0) beta[t] = red[0];
}

__global__ void softmax_kernel(const float* __restrict__ beta, float* __restrict__ alfa) {
  __shared__ float red[256];
  float m = -INFINITY;
  for (int t = threadIdx.x; t < L_SEQ; t += 256) m = fmaxf(m, beta[t]);
  red[threadIdx.x] = m; __syncthreads();
  for (int s = 128; s > 0; s >>= 1) {
    if (threadIdx.x < s) red[threadIdx.x] = fmaxf(red[threadIdx.x], red[threadIdx.x + s]);
    __syncthreads();
  }
  float mx = red[0]; __syncthreads();
  float sum = 0.f;
  for (int t = threadIdx.x; t < L_SEQ; t += 256) sum += __expf(beta[t] - mx);
  red[threadIdx.x] = sum; __syncthreads();
  for (int s = 128; s > 0; s >>= 1) {
    if (threadIdx.x < s) red[threadIdx.x] += red[threadIdx.x + s];
    __syncthreads();
  }
  float inv = 1.0f / red[0];
  for (int t = threadIdx.x; t < L_SEQ; t += 256) alfa[t] = __expf(beta[t] - mx) * inv;
}

__global__ void wsum_kernel(const float* __restrict__ alfa, const float* __restrict__ prod,
                            float* __restrict__ s) {
  int j = (int)blockIdx.x * 256 + threadIdx.x;
  float acc = 0.f;
  for (int t = 0; t < L_SEQ; ++t) acc += alfa[t] * prod[(size_t)t * H_DIM + j];
  s[j] = acc;
}

__global__ void final_kernel(const float* __restrict__ s, const float* __restrict__ lin2_w,
                             const float* __restrict__ lin2_b, float* __restrict__ out) {
  int l = threadIdx.x >> 5, lane = threadIdx.x & 31;
  if (l < 3) {
    float acc = 0.f;
    for (int j = lane; j < H_DIM; j += 32) acc += s[j] * lin2_w[(size_t)l * H_DIM + j];
    for (int off = 16; off > 0; off >>= 1) acc += __shfl_down(acc, off);
    if (lane == 0) out[l] = acc + lin2_b[l];
  }
}

// ---------------------------------------------------------------------------
extern "C" void kernel_launch(void* const* d_in, const int* in_sizes, int n_in,
                              void* d_out, int out_size, void* d_ws, size_t ws_size,
                              hipStream_t stream) {
  const int*   x      = (const int*)d_in[0];
  const int*   ts     = (const int*)d_in[1];
  const int*   te     = (const int*)d_in[2];
  const float* emb    = (const float*)d_in[3];
  const float* w_ih_l = (const float*)d_in[4];
  const float* w_hh_l = (const float*)d_in[5];
  const float* b_ih_l = (const float*)d_in[6];
  const float* b_hh_l = (const float*)d_in[7];
  const float* w_ih_r = (const float*)d_in[8];
  const float* w_hh_r = (const float*)d_in[9];
  const float* b_ih_r = (const float*)d_in[10];
  const float* b_hh_r = (const float*)d_in[11];
  const float* lin1_w = (const float*)d_in[12];
  const float* lin1_b = (const float*)d_in[13];
  const float* u      = (const float*)d_in[14];
  const float* lin2_w = (const float*)d_in[15];
  const float* lin2_b = (const float*)d_in[16];
  float* out = (float*)d_out;

  char* w = (char*)d_ws;
  size_t off = 0;
  auto alloc = [&](size_t bytes) -> char* {
    char* p = w + off; off += (bytes + 255) & ~(size_t)255; return p;
  };
  float*          target  = (float*)alloc(E_DIM * 4);
  float*          e_l     = (float*)alloc(E_DIM * 4);
  float*          e_r     = (float*)alloc(E_DIM * 4);
  float*          gbase_l = (float*)alloc(G4 * 4);
  float*          gspec_l = (float*)alloc(G4 * 4);
  float*          gbase_r = (float*)alloc(G4 * 4);
  float*          gspec_r = (float*)alloc(G4 * 4);
  int*            meta    = (int*)alloc(64);
  unsigned int*   ctr     = (unsigned int*)alloc(256);
  float*          h_g     = (float*)alloc(H_DIM * 4);
  float*          c_g     = (float*)alloc(H_DIM * 4);
  unsigned short* pk_l    = (unsigned short*)alloc((size_t)G4 * H_DIM * 2);
  unsigned short* pk_r    = (unsigned short*)alloc((size_t)G4 * H_DIM * 2);
  float*          hs_l    = (float*)alloc((size_t)L_SEQ * H_DIM * 4);
  float*          hs_r    = (float*)alloc((size_t)L_SEQ * H_DIM * 4);
  float*          prod    = (float*)alloc((size_t)L_SEQ * H_DIM * 4);
  unsigned short* pA      = (unsigned short*)alloc((size_t)L_SEQ * H_DIM * 2);
  unsigned short* pB      = (unsigned short*)alloc((size_t)H_DIM * H_DIM * 2);
  float*          t_out   = (float*)alloc((size_t)L_SEQ * H_DIM * 4);
  float*          beta    = (float*)alloc(L_SEQ * 4);
  float*          alfa    = (float*)alloc(L_SEQ * 4);
  float*          sVec    = (float*)alloc(H_DIM * 4);

  (void)hipFuncSetAttribute((const void*)lstm_pass_kernel,
                            hipFuncAttributeMaxDynamicSharedMemorySize, DYN_LDS);

  (void)hipMemsetAsync(ctr, 0, 256, stream);
  (void)hipMemsetAsync(h_g, 0, H_DIM * 4, stream);

  prep_kernel<<<1, 256, 0, stream>>>(x, ts, te, emb, target, e_l, e_r, meta);

  base_gates_kernel<<<dim3(G4, 2), 128, 0, stream>>>(
      w_ih_l, b_ih_l, b_hh_l, w_ih_r, b_ih_r, b_hh_r,
      target, e_l, e_r, gbase_l, gspec_l, gbase_r, gspec_r);

  pack_whh_kernel<<<dim3((G4 * H_DIM) / 256, 2), 256, 0, stream>>>(w_hh_l, w_hh_r, pk_l, pk_r);

  lstm_pass_kernel<<<NB, 256, DYN_LDS, stream>>>(pk_l, gbase_l, gspec_l, meta, 0,
                                                 hs_l, h_g, c_g, ctr + 0, 0);
  lstm_pass_kernel<<<NB, 256, DYN_LDS, stream>>>(pk_r, gbase_r, gspec_r, meta, 1,
                                                 hs_r, h_g, c_g, ctr + 1, 1);

  prod_pack_kernel<<<(L_SEQ * H_DIM) / 256, 256, 0, stream>>>(hs_l, hs_r, prod, pA);
  pack_lin1_kernel<<<(H_DIM * H_DIM) / 256, 256, 0, stream>>>(lin1_w, pB);

  gemm_tanh_kernel<<<1024, 256, 0, stream>>>(pA, pB, lin1_b, t_out);

  beta_kernel<<<L_SEQ, 256, 0, stream>>>(t_out, u, beta);
  softmax_kernel<<<1, 256, 0, stream>>>(beta, alfa);
  wsum_kernel<<<H_DIM / 256, 256, 0, stream>>>(alfa, prod, sVec);
  final_kernel<<<1, 128, 0, stream>>>(sVec, lin2_w, lin2_b, out);
}
